// HackyActorCritic_38878043964064
// MI455X (gfx1250) — compile-verified, matrix operand order checked by reading, not run on hardware
//
#include <hip/hip_runtime.h>

typedef __attribute__((ext_vector_type(16))) _Float16 v16h;
typedef __attribute__((ext_vector_type(8)))  _Float16 v8h;
typedef __attribute__((ext_vector_type(8)))  float    v8f;

union HF16 { v16h f; v8h h[2]; };

#define D0 795
#define KC 416            // K staging chunk (multiple of 32)
#define K1 832            // D0 padded: 2 * KC
#define D1 512
#define D2 256
#define D3 128
#define D4 14
#define BM 64             // batch rows per block
#define BATCH 65536

// f16 weight workspace layout (element offsets)
#define W1_OFF 0
#define W2_OFF (D1 * K1)                 // 425984
#define W3_OFF (W2_OFF + D2 * D1)        // 557056
#define W4_OFF (W3_OFF + D3 * D2)        // 589824
#define W_TOTAL (W4_OFF + 16 * D3)       // 591872 halves = 1183744 bytes

__device__ inline v8f vzero() {
    v8f z;
#pragma unroll
    for (int j = 0; j < 8; ++j) z[j] = 0.f;
    return z;
}

__device__ inline HF16 load_frag(const _Float16* p) {
    HF16 f;
    f.h[0] = *(const v8h*)(p);
    f.h[1] = *(const v8h*)(p + 16);
    return f;
}

__device__ inline v8f wmma_f16(const HF16& a, const HF16& b, v8f c) {
    return __builtin_amdgcn_wmma_f32_16x16x32_f16(
        false, a.f, false, b.f, (short)0, c, false, false);
}

__device__ inline float elu(float v) { return (v < 0.f) ? (__expf(v) - 1.f) : v; }

// load the 4 A fragments (one per M-tile) for one 32-wide k step
__device__ inline void loadA4(HF16 (&a)[4], const _Float16* __restrict__ sIn,
                              int stride, int lm, int kcol) {
#pragma unroll
    for (int mt = 0; mt < 4; ++mt)
        a[mt] = load_frag(sIn + (mt * 16 + lm) * stride + kcol);
}

template <int NT>
__device__ inline void loadB(HF16 (&b)[NT], const _Float16* __restrict__ wG,
                             int stride, int nbase, int lm, int kcol) {
#pragma unroll
    for (int nt = 0; nt < NT; ++nt)
        b[nt] = load_frag(wG + (size_t)(nbase + nt * 16 + lm) * stride + kcol);
}

template <int NT>
__device__ inline void mma_group(v8f (&acc)[4][NT], const HF16 (&a)[4], const HF16 (&b)[NT]) {
#pragma unroll
    for (int nt = 0; nt < NT; ++nt)
#pragma unroll
        for (int mt = 0; mt < 4; ++mt)
            acc[mt][nt] = wmma_f16(a[mt], b[nt], acc[mt][nt]);
}

// ---------------- weight conversion: f32 -> padded f16 -------------------
__global__ void convert_weights(const float* __restrict__ W1,
                                const float* __restrict__ W2,
                                const float* __restrict__ W3,
                                const float* __restrict__ W4,
                                _Float16* __restrict__ wf) {
    int stride = gridDim.x * blockDim.x;
    for (int i = blockIdx.x * blockDim.x + threadIdx.x; i < W_TOTAL; i += stride) {
        float v;
        if (i < W2_OFF) {
            int n = i / K1, k = i - n * K1;
            v = (k < D0) ? W1[n * D0 + k] : 0.f;
        } else if (i < W3_OFF) {
            int j = i - W2_OFF; int n = j / D1, k = j - n * D1;
            v = W2[n * D1 + k];
        } else if (i < W4_OFF) {
            int j = i - W3_OFF; int n = j / D2, k = j - n * D2;
            v = W3[n * D2 + k];
        } else {
            int j = i - W4_OFF; int n = j / D3, k = j - n * D3;
            v = (n < D4) ? W4[n * D3 + k] : 0.f;
        }
        wf[i] = (_Float16)v;
    }
}

// LDS-input layer with ping-pong (double-buffered) B fragments.
// out[64 x ldOut] = elu(in[64 x K] @ W^T + b); each wave: 4 M-tiles x NT N-tiles
template <int K, int NT>
__device__ inline void layer_lds(const _Float16* __restrict__ sIn,
                                 const _Float16* __restrict__ wG,
                                 const float* __restrict__ bias,
                                 _Float16* __restrict__ sOut, int ldOut,
                                 int nbase, int lm, int half) {
    constexpr int STEPS = K / 32;
    const int koff = half * 8;
    v8f acc[4][NT];
#pragma unroll
    for (int mt = 0; mt < 4; ++mt)
#pragma unroll
        for (int nt = 0; nt < NT; ++nt) acc[mt][nt] = vzero();

    HF16 a[4], b0[NT], b1[NT];
    loadB<NT>(b0, wG, K, nbase, lm, koff);
    int s = 0;
    for (; s + 2 <= STEPS; s += 2) {
        loadB<NT>(b1, wG, K, nbase, lm, (s + 1) * 32 + koff);
        loadA4(a, sIn, K, lm, s * 32 + koff);
        mma_group<NT>(acc, a, b0);
        if (s + 2 < STEPS)
            loadB<NT>(b0, wG, K, nbase, lm, (s + 2) * 32 + koff);
        loadA4(a, sIn, K, lm, (s + 1) * 32 + koff);
        mma_group<NT>(acc, a, b1);
    }
    if (s < STEPS) {            // odd tail
        loadA4(a, sIn, K, lm, s * 32 + koff);
        mma_group<NT>(acc, a, b0);
    }
#pragma unroll
    for (int nt = 0; nt < NT; ++nt) {
        int n = nbase + nt * 16 + lm;
        float bv = bias[n];
#pragma unroll
        for (int mt = 0; mt < 4; ++mt)
#pragma unroll
            for (int v = 0; v < 8; ++v) {
                int m = mt * 16 + v + half * 8;
                sOut[m * ldOut + n] = (_Float16)elu(acc[mt][nt][v] + bv);
            }
    }
}

// --------------------------- fused MLP -----------------------------------
__global__ __launch_bounds__(256, 1)
void mlp_fused(const float* __restrict__ x,
               const _Float16* __restrict__ wf,
               const float* __restrict__ b1, const float* __restrict__ b2,
               const float* __restrict__ b3, const float* __restrict__ b4,
               float* __restrict__ out) {
    __shared__ __align__(16) _Float16 smemA[BM * KC];  // xs staging, later h2
    __shared__ __align__(16) _Float16 smemB[BM * D1];  // h1, later h3

    _Float16* xs = smemA;
    _Float16* h1 = smemB;
    _Float16* h2 = smemA;   // xs dead after layer 1
    _Float16* h3 = smemB;   // h1 dead after layer 2

    const int tid  = threadIdx.x;
    const int lane = tid & 31;
    const int wv   = tid >> 5;    // 0..7: N-slice owner
    const int lm   = lane & 15;
    const int half = lane >> 4;
    const int koff = half * 8;
    const int row0 = blockIdx.x * BM;

    // ---- Layer 1: x(64x795) @ W1^T -> h1(64x512), K staged in 2 phases ----
    {
        constexpr int STEPS = KC / 32;   // 13
        v8f acc[4][4];
#pragma unroll
        for (int mt = 0; mt < 4; ++mt)
#pragma unroll
            for (int nt = 0; nt < 4; ++nt) acc[mt][nt] = vzero();

        const int nbase = wv * 64;
        const _Float16* w1 = wf + W1_OFF;
        for (int phase = 0; phase < 2; ++phase) {
            __syncthreads();   // previous-phase reads complete before restage
            for (int idx = tid; idx < BM * KC; idx += 256) {
                int r = idx / KC, c = idx - r * KC;
                int gc = phase * KC + c;
                float v = (gc < D0) ? x[(size_t)(row0 + r) * D0 + gc] : 0.f;
                xs[idx] = (_Float16)v;
            }
            __syncthreads();
            const int kw = phase * KC;

            HF16 a[4], b0[4], b1f[4];
            loadB<4>(b0, w1, K1, nbase, lm, kw + koff);
            int s = 0;
            for (; s + 2 <= STEPS; s += 2) {
                loadB<4>(b1f, w1, K1, nbase, lm, kw + (s + 1) * 32 + koff);
                loadA4(a, xs, KC, lm, s * 32 + koff);
                mma_group<4>(acc, a, b0);
                if (s + 2 < STEPS)
                    loadB<4>(b0, w1, K1, nbase, lm, kw + (s + 2) * 32 + koff);
                loadA4(a, xs, KC, lm, (s + 1) * 32 + koff);
                mma_group<4>(acc, a, b1f);
            }
            if (s < STEPS) {    // odd tail (STEPS = 13)
                loadA4(a, xs, KC, lm, s * 32 + koff);
                mma_group<4>(acc, a, b0);
            }
        }
        __syncthreads();
#pragma unroll
        for (int nt = 0; nt < 4; ++nt) {
            int n = nbase + nt * 16 + lm;
            float bv = b1[n];
#pragma unroll
            for (int mt = 0; mt < 4; ++mt)
#pragma unroll
                for (int v = 0; v < 8; ++v) {
                    int m = mt * 16 + v + half * 8;
                    h1[m * D1 + n] = (_Float16)elu(acc[mt][nt][v] + bv);
                }
        }
    }
    __syncthreads();

    // ---- Layer 2: h1(64x512) -> h2(64x256) ----
    layer_lds<D1, 2>(h1, wf + W2_OFF, b2, h2, D2, wv * 32, lm, half);
    __syncthreads();

    // ---- Layer 3: h2(64x256) -> h3(64x128) ----
    layer_lds<D2, 1>(h2, wf + W3_OFF, b3, h3, D3, wv * 16, lm, half);
    __syncthreads();

    // ---- Layer 4: h3(64x128) -> out(64x14), no ELU ----
    if (wv == 0) {   // wave-uniform branch: EXEC all-ones inside
        v8f acc[4];
#pragma unroll
        for (int mt = 0; mt < 4; ++mt) acc[mt] = vzero();
        for (int k0 = 0; k0 < D3; k0 += 32) {
            HF16 a[4];
            loadA4(a, h3, D3, lm, k0 + koff);
            HF16 b = load_frag(wf + W4_OFF + (size_t)lm * D3 + k0 + koff);
#pragma unroll
            for (int mt = 0; mt < 4; ++mt)
                acc[mt] = wmma_f16(a[mt], b, acc[mt]);
        }
        if (lm < D4) {
            float bv = b4[lm];
#pragma unroll
            for (int mt = 0; mt < 4; ++mt)
#pragma unroll
                for (int v = 0; v < 8; ++v) {
                    int mg = row0 + mt * 16 + v + half * 8;
                    out[(size_t)mg * D4 + lm] = acc[mt][v] + bv;
                }
        }
    }
}

extern "C" void kernel_launch(void* const* d_in, const int* in_sizes, int n_in,
                              void* d_out, int out_size, void* d_ws, size_t ws_size,
                              hipStream_t stream) {
    const float* x  = (const float*)d_in[0];
    const float* W1 = (const float*)d_in[1];
    const float* b1 = (const float*)d_in[2];
    const float* W2 = (const float*)d_in[3];
    const float* b2 = (const float*)d_in[4];
    const float* W3 = (const float*)d_in[5];
    const float* b3 = (const float*)d_in[6];
    const float* W4 = (const float*)d_in[7];
    const float* b4 = (const float*)d_in[8];
    _Float16* wf = (_Float16*)d_ws;

    convert_weights<<<512, 256, 0, stream>>>(W1, W2, W3, W4, wf);
    mlp_fused<<<BATCH / BM, 256, 0, stream>>>(x, wf, b1, b2, b3, b4, (float*)d_out);
}
